// FuzzyHesitantEnsemble_24223615550231
// MI455X (gfx1250) — compile-verified, matrix operand order checked by reading, not run on hardware
//
#include <hip/hip_runtime.h>
#include <hip/hip_bf16.h>

typedef __attribute__((ext_vector_type(16))) _Float16 v16h;
typedef __attribute__((ext_vector_type(8)))  _Float16 v8h;
typedef __attribute__((ext_vector_type(8)))  float    v8f;

#define B_   256
#define M_   8
#define K_   3

// ---------------- workspace layout (bytes, all 256B aligned) ----------------
static constexpr size_t SZ_H1  = (size_t)256*112*112*32*2;   // 205,520,896
static constexpr size_t SZ_H2  = (size_t)256*56*56*64*2;     // 102,760,448
static constexpr size_t O_H1   = 0;
static constexpr size_t O_H2   = O_H1  + SZ_H1;
static constexpr size_t O_WP2  = O_H2  + SZ_H2;              // 9*4*1*512*2   = 36,864
static constexpr size_t O_WP3  = O_WP2 + 36864;              // 9*8*2*512*2   = 147,456
static constexpr size_t O_WL1  = O_WP3 + 147456;             // 8*4*512*2     = 32,768
static constexpr size_t O_WL2  = O_WL1 + 32768;              // 2*4*512*2     = 8,192
static constexpr size_t O_WPM  = O_WL2 + 8192;               // 8*3*224*224*4 = 4,816,896
static constexpr size_t O_S    = O_WPM + 4816896;            // 24 f32 -> pad 256
static constexpr size_t O_FEATS= O_S   + 256;                // 256*128*4     = 131,072
static constexpr size_t O_HID  = O_FEATS + 131072;           // 256*128*4     = 131,072
static constexpr size_t O_MEM  = O_HID + 131072;             // 256*24*4      = 24,576
static constexpr size_t O_OUTR = O_MEM + 24576;              // 256*8*4       = 8,192

// K-index a lane's j-th f16 slot maps to, per the documented 16-bit A/B
// fragment layout (lanes 0-15: K {0..7, 16..23}; lanes 16-31: K {8..15, 24..31}).
__device__ __forceinline__ int kmap(int lane, int j) {
    int base = (lane >= 16) ? 8 : 0;
    return (j < 8) ? (base + j) : (16 + base + (j - 8));
}

__device__ __forceinline__ v8f wmma_f16(v16h a, v16h b, v8f c) {
    return __builtin_amdgcn_wmma_f32_16x16x32_f16(false, a, false, b, (short)0, c, false, false);
}

// ---------------- weight pre-pack kernels (B fragments, BN scale folded) ----
// layout: (((tap*CT + ct)*KS + ks)*32 + lane)*16 + j
__global__ void k_pack_conv(const float* __restrict__ w, const float* __restrict__ scale,
                            _Float16* __restrict__ dst, int CT, int KS) {
    int tid = blockIdx.x * 256 + threadIdx.x;
    int total = 9 * CT * KS * 512;
    if (tid >= total) return;
    int j = tid & 15, lane = (tid >> 4) & 31, q = tid >> 9;
    int ks = q % KS; q /= KS;
    int ct = q % CT; int tap = q / CT;
    int Cin = KS * 32;
    int ci = ks * 32 + kmap(lane, j);
    int co = ct * 16 + (lane & 15);
    float v = w[(size_t)(co * Cin + ci) * 9 + tap] * scale[co];
    dst[tid] = (_Float16)v;
}

// layout: ((nt*KS + ks)*32 + lane)*16 + j ; B[k][n] = w[n*K + k]
__global__ void k_pack_linear(const float* __restrict__ w, _Float16* __restrict__ dst,
                              int N, int Ktot) {
    int tid = blockIdx.x * 256 + threadIdx.x;
    int KS = Ktot / 32, NT = (N + 15) / 16;
    int total = NT * KS * 512;
    if (tid >= total) return;
    int j = tid & 15, lane = (tid >> 4) & 31, q = tid >> 9;
    int ks = q % KS; int nt = q / KS;
    int k = ks * 32 + kmap(lane, j);
    int n = nt * 16 + (lane & 15);
    dst[tid] = (n < N) ? (_Float16)w[(size_t)n * Ktot + k] : (_Float16)0.f;
}

// ---------------- conv1: direct f32 VALU, out = f16 NHWC [B,112,112,32] -----
__global__ __launch_bounds__(256) void k_conv1(const float* __restrict__ x,
                                               const float* __restrict__ w,
                                               const float* __restrict__ scale,
                                               const float* __restrict__ bias,
                                               _Float16* __restrict__ h1) {
    __shared__ float lw[928];                       // 864 w + 32 scale + 32 bias
    for (int i = threadIdx.x; i < 928; i += 256)
        lw[i] = (i < 864) ? w[i] : (i < 896 ? scale[i - 864] : bias[i - 896]);
    __syncthreads();
    int tid = blockIdx.x * 256 + threadIdx.x;       // 256*112*112 positions
    int b = tid / 12544, rem = tid % 12544;
    int oy = rem / 112, ox = rem % 112;
    float xv[27];
    #pragma unroll
    for (int c = 0; c < 3; ++c)
        #pragma unroll
        for (int t = 0; t < 9; ++t) {
            int iy = 2 * oy - 1 + t / 3, ix = 2 * ox - 1 + t % 3;
            bool ok = (iy >= 0) & (iy < 224) & (ix >= 0) & (ix < 224);
            xv[c * 9 + t] = ok ? x[(((size_t)b * 3 + c) * 224 + iy) * 224 + ix] : 0.f;
        }
    #pragma unroll
    for (int co = 0; co < 32; ++co) {
        float a = 0.f;
        #pragma unroll
        for (int q = 0; q < 27; ++q) a += xv[q] * lw[co * 27 + q];
        a = a * lw[864 + co] + lw[896 + co];
        h1[(size_t)tid * 32 + co] = (_Float16)fmaxf(a, 0.f);
    }
}

// ---------------- conv2: WMMA GEMM, K = 9 taps x 32ch, out [B,56,56,64] -----
__global__ __launch_bounds__(128) void k_conv2(const _Float16* __restrict__ h1,
                                               const _Float16* __restrict__ wp,
                                               const float* __restrict__ bias,
                                               _Float16* __restrict__ h2) {
    const int lane = threadIdx.x & 31, wave = threadIdx.x >> 5;
    const int b = blockIdx.y;
    const int pbase = blockIdx.x * 64 + wave * 16;  // 49*64 = 3136 positions
    const bool hi = lane >= 16;
    const int p = pbase + (lane & 15);
    const int oy = p / 56, ox = p % 56;
    const int khalf = hi ? 8 : 0;

    v8f acc[4];
    #pragma unroll
    for (int t = 0; t < 4; ++t)
        #pragma unroll
        for (int r = 0; r < 8; ++r) acc[t][r] = 0.f;

    #pragma unroll
    for (int tap = 0; tap < 9; ++tap) {
        int iy = 2 * oy - 1 + tap / 3, ix = 2 * ox - 1 + tap % 3;
        v16h a;
        #pragma unroll
        for (int q = 0; q < 16; ++q) a[q] = (_Float16)0.f;
        if (iy >= 0 && iy < 112 && ix >= 0 && ix < 112) {
            const _Float16* rp = h1 + (((size_t)b * 112 + iy) * 112 + ix) * 32 + khalf;
            v8h lo = *(const v8h*)rp;
            v8h hh = *(const v8h*)(rp + 16);
            #pragma unroll
            for (int q = 0; q < 8; ++q) { a[q] = lo[q]; a[8 + q] = hh[q]; }
        }
        #pragma unroll
        for (int t = 0; t < 4; ++t) {
            v16h bf = *(const v16h*)(wp + ((size_t)(tap * 4 + t) * 32 + lane) * 16);
            acc[t] = wmma_f16(a, bf, acc[t]);
        }
    }
    #pragma unroll
    for (int r = 0; r < 8; ++r) {
        int pp = pbase + r + (hi ? 8 : 0);
        size_t orow = ((size_t)b * 3136 + pp) * 64;
        #pragma unroll
        for (int t = 0; t < 4; ++t) {
            int co = t * 16 + (lane & 15);
            float v = acc[t][r] + bias[co];
            h2[orow + co] = (_Float16)fmaxf(v, 0.f);
        }
    }
}

// ------- conv3: WMMA GEMM (K = 9 taps x 64ch) fused with global avg pool ----
__global__ __launch_bounds__(128) void k_conv3(const _Float16* __restrict__ h2,
                                               const _Float16* __restrict__ wp,
                                               const float* __restrict__ bias,
                                               float* __restrict__ feats) {
    __shared__ float fl[128];
    fl[threadIdx.x] = 0.f;
    __syncthreads();
    const int lane = threadIdx.x & 31, wave = threadIdx.x >> 5;
    const int b = blockIdx.y;
    const int pbase = blockIdx.x * 64 + wave * 16;  // 13 blocks cover 784 (masked)
    const bool hi = lane >= 16;
    const int p = pbase + (lane & 15);
    const int oy = p / 28, ox = p % 28;
    const int khalf = hi ? 8 : 0;
    const bool prow_ok = p < 784;

    v8f acc[8];
    #pragma unroll
    for (int t = 0; t < 8; ++t)
        #pragma unroll
        for (int r = 0; r < 8; ++r) acc[t][r] = 0.f;

    #pragma unroll
    for (int tap = 0; tap < 9; ++tap) {
        int iy = 2 * oy - 1 + tap / 3, ix = 2 * ox - 1 + tap % 3;
        bool ok = prow_ok && iy >= 0 && iy < 56 && ix >= 0 && ix < 56;
        const _Float16* rbase = h2 + (((size_t)b * 56 + iy) * 56 + ix) * 64;
        #pragma unroll
        for (int ks = 0; ks < 2; ++ks) {
            v16h a;
            #pragma unroll
            for (int q = 0; q < 16; ++q) a[q] = (_Float16)0.f;
            if (ok) {
                const _Float16* rp = rbase + ks * 32 + khalf;
                v8h lo = *(const v8h*)rp;
                v8h hh = *(const v8h*)(rp + 16);
                #pragma unroll
                for (int q = 0; q < 8; ++q) { a[q] = lo[q]; a[8 + q] = hh[q]; }
            }
            #pragma unroll
            for (int t = 0; t < 8; ++t) {
                v16h bf = *(const v16h*)(wp + ((size_t)((tap * 8 + t) * 2 + ks) * 32 + lane) * 16);
                acc[t] = wmma_f16(a, bf, acc[t]);
            }
        }
    }
    #pragma unroll
    for (int r = 0; r < 8; ++r) {
        int pp = pbase + r + (hi ? 8 : 0);
        if (pp < 784) {
            #pragma unroll
            for (int t = 0; t < 8; ++t) {
                int co = t * 16 + (lane & 15);
                float v = fmaxf(acc[t][r] + bias[co], 0.f);
                atomicAdd(&fl[co], v);              // ds_add_f32
            }
        }
    }
    __syncthreads();
    atomicAdd(&feats[(size_t)b * 128 + threadIdx.x], fl[threadIdx.x]);
}

// ---------------- lin1: WMMA [256,128]x[128,128], relu ----------------------
__global__ __launch_bounds__(256) void k_lin1(const float* __restrict__ feats,
                                              const _Float16* __restrict__ wp,
                                              const float* __restrict__ bias,
                                              float* __restrict__ hid) {
    const int lane = threadIdx.x & 31, nt = threadIdx.x >> 5;  // 8 waves = 8 N-tiles
    const int mt = blockIdx.x;                                 // 16 M-tiles
    const bool hi = lane >= 16;
    const int mrow = mt * 16 + (lane & 15);
    v8f acc;
    #pragma unroll
    for (int r = 0; r < 8; ++r) acc[r] = 0.f;
    #pragma unroll
    for (int ks = 0; ks < 4; ++ks) {
        const float* fr = feats + (size_t)mrow * 128 + ks * 32 + (hi ? 8 : 0);
        v16h a;
        #pragma unroll
        for (int q = 0; q < 8; ++q) {
            a[q]     = (_Float16)(fr[q]      * (1.f / 784.f));   // mean pool fold
            a[8 + q] = (_Float16)(fr[16 + q] * (1.f / 784.f));
        }
        v16h bf = *(const v16h*)(wp + ((size_t)(nt * 4 + ks) * 32 + lane) * 16);
        acc = wmma_f16(a, bf, acc);
    }
    #pragma unroll
    for (int r = 0; r < 8; ++r) {
        int m = mt * 16 + r + (hi ? 8 : 0);
        int n = nt * 16 + (lane & 15);
        hid[(size_t)m * 128 + n] = fmaxf(acc[r] + bias[n], 0.f);
    }
}

// ---------------- lin2: WMMA [256,128]x[128,24] -----------------------------
__global__ __launch_bounds__(64) void k_lin2(const float* __restrict__ hid,
                                             const _Float16* __restrict__ wp,
                                             const float* __restrict__ bias,
                                             float* __restrict__ mem_pre) {
    const int lane = threadIdx.x & 31, nt = threadIdx.x >> 5;  // 2 waves
    const int mt = blockIdx.x;
    const bool hi = lane >= 16;
    const int mrow = mt * 16 + (lane & 15);
    v8f acc;
    #pragma unroll
    for (int r = 0; r < 8; ++r) acc[r] = 0.f;
    #pragma unroll
    for (int ks = 0; ks < 4; ++ks) {
        const float* fr = hid + (size_t)mrow * 128 + ks * 32 + (hi ? 8 : 0);
        v16h a;
        #pragma unroll
        for (int q = 0; q < 8; ++q) { a[q] = (_Float16)fr[q]; a[8 + q] = (_Float16)fr[16 + q]; }
        v16h bf = *(const v16h*)(wp + ((size_t)(nt * 4 + ks) * 32 + lane) * 16);
        acc = wmma_f16(a, bf, acc);
    }
    int n = nt * 16 + (lane & 15);
    if (n < 24) {
        #pragma unroll
        for (int r = 0; r < 8; ++r) {
            int m = mt * 16 + r + (hi ? 8 : 0);
            mem_pre[(size_t)m * 24 + n] = acc[r] + bias[n];
        }
    }
}

// ---------------- sub-model prep: Wp = w/std ; S[m,c] = sum_hw w ------------
__global__ void k_wp(const float* __restrict__ mw, const float* __restrict__ stds,
                     float* __restrict__ wpm) {
    int i = blockIdx.x * 256 + threadIdx.x;           // 1,204,224 exact
    int m = i / 150528, c = (i / 50176) % 3;
    wpm[i] = mw[i] / stds[m * 3 + c];
}
__global__ __launch_bounds__(256) void k_wsum(const float* __restrict__ mw,
                                              float* __restrict__ S) {
    int mc = blockIdx.x;                              // 24 blocks
    const float* p = mw + (size_t)mc * 50176;
    float s = 0.f;
    for (int i = threadIdx.x; i < 50176; i += 256) s += p[i];
    __shared__ float r[256];
    r[threadIdx.x] = s; __syncthreads();
    for (int st = 128; st; st >>= 1) {
        if (threadIdx.x < st) r[threadIdx.x] += r[threadIdx.x + st];
        __syncthreads();
    }
    if (threadIdx.x == 0) S[mc] = r[0];
}

// ---------------- outputs einsum: streaming f32, 8 b-rows x 8 models --------
__global__ __launch_bounds__(256) void k_outputs(const float* __restrict__ x,
                                                 const float* __restrict__ wpm,
                                                 float* __restrict__ outr) {
    const int bt = blockIdx.x;                        // 32 batch tiles of 8
    const int ch = blockIdx.y;                        // 12 K-chunks of 12544
    const int kend = (ch + 1) * 12544;
    const float* xb = x + (size_t)bt * 8 * 150528;
    float acc[8][8];
    #pragma unroll
    for (int i = 0; i < 8; ++i)
        #pragma unroll
        for (int m = 0; m < 8; ++m) acc[i][m] = 0.f;
    for (int k = ch * 12544 + (int)threadIdx.x; k < kend; k += 256) {
        __builtin_prefetch(&xb[k + 2048], 0, 1);      // global_prefetch_b8
        float wv[8];
        #pragma unroll
        for (int m = 0; m < 8; ++m) wv[m] = wpm[(size_t)m * 150528 + k];
        #pragma unroll
        for (int i = 0; i < 8; ++i) {
            float xv = xb[(size_t)i * 150528 + k];
            #pragma unroll
            for (int m = 0; m < 8; ++m) acc[i][m] += xv * wv[m];
        }
    }
    __shared__ float red[64];
    if (threadIdx.x < 64) red[threadIdx.x] = 0.f;
    __syncthreads();
    #pragma unroll
    for (int i = 0; i < 8; ++i)
        #pragma unroll
        for (int m = 0; m < 8; ++m) atomicAdd(&red[i * 8 + m], acc[i][m]);
    __syncthreads();
    if (threadIdx.x < 64) atomicAdd(&outr[bt * 64 + threadIdx.x], red[threadIdx.x]);
}

// ---------------- gating / fuzzy top-p / final combine ----------------------
__global__ __launch_bounds__(256) void k_combine(const float* __restrict__ mem_pre,
                                                 const float* __restrict__ aggw,
                                                 const float* __restrict__ outr,
                                                 const float* __restrict__ model_b,
                                                 const float* __restrict__ means,
                                                 const float* __restrict__ stds,
                                                 const float* __restrict__ S,
                                                 float* __restrict__ out) {
    int b = threadIdx.x;                              // 1 block of 256
    // agg softmax (K=3)
    float a0 = aggw[0], a1 = aggw[1], a2 = aggw[2];
    float mx = fmaxf(a0, fmaxf(a1, a2));
    float e0 = __expf(a0 - mx), e1 = __expf(a1 - mx), e2 = __expf(a2 - mx);
    float es = e0 + e1 + e2;
    float g0 = e0 / es, g1 = e1 / es, g2 = e2 / es;

    float fw[8]; float hes_sum = 0.f;
    #pragma unroll
    for (int m = 0; m < 8; ++m) {
        float s0 = 1.f / (1.f + __expf(-mem_pre[b * 24 + m * 3 + 0]));
        float s1 = 1.f / (1.f + __expf(-mem_pre[b * 24 + m * 3 + 1]));
        float s2 = 1.f / (1.f + __expf(-mem_pre[b * 24 + m * 3 + 2]));
        float mn = (s0 + s1 + s2) * (1.f / 3.f);
        hes_sum += ((s0 - mn) * (s0 - mn) + (s1 - mn) * (s1 - mn) + (s2 - mn) * (s2 - mn)) * 0.5f;
        fw[m] = s0 * g0 + s1 * g1 + s2 * g2;
    }
    float wmx = fw[0];
    #pragma unroll
    for (int m = 1; m < 8; ++m) wmx = fmaxf(wmx, fw[m]);
    float wsum = 0.f;
    #pragma unroll
    for (int m = 0; m < 8; ++m) { fw[m] = __expf(fw[m] - wmx); wsum += fw[m]; }
    #pragma unroll
    for (int m = 0; m < 8; ++m) fw[m] /= wsum;
    float avg_hes = hes_sum * (1.f / 8.f);

    // descending sort of 8 weights with index tracking
    int ord[8] = {0, 1, 2, 3, 4, 5, 6, 7};
    #pragma unroll
    for (int i = 0; i < 7; ++i)
        #pragma unroll
        for (int j2 = 0; j2 < 7; ++j2)
            if (fw[ord[j2 + 1]] > fw[ord[j2]]) { int t = ord[j2]; ord[j2] = ord[j2 + 1]; ord[j2 + 1] = t; }

    float mask[8];
    float cum = 0.f;
    #pragma unroll
    for (int i = 0; i < 8; ++i) { mask[ord[i]] = (cum < 0.9f) ? 1.f : 0.f; cum += fw[ord[i]]; }
    if (avg_hes > 0.2f) {
        #pragma unroll
        for (int m = 0; m < 8; ++m) mask[m] = 1.f;
    }
    float rs = 1e-8f;
    #pragma unroll
    for (int m = 0; m < 8; ++m) rs += fw[m] * mask[m];

    float fo = 0.f;
    #pragma unroll
    for (int m = 0; m < 8; ++m) {
        float beff = model_b[m];
        #pragma unroll
        for (int c = 0; c < 3; ++c)
            beff -= means[m * 3 + c] * S[m * 3 + c] / stds[m * 3 + c];
        float w = fw[m] * mask[m] / rs;
        fo += w * (outr[b * 8 + m] + beff);
        out[256 + b * 8 + m] = w;                     // fw output [B,8]
    }
    out[b] = fo;                                      // final_output [B,1]
}

// ---------------------------------------------------------------------------
extern "C" void kernel_launch(void* const* d_in, const int* in_sizes, int n_in,
                              void* d_out, int out_size, void* d_ws, size_t ws_size,
                              hipStream_t stream) {
    const float* x        = (const float*)d_in[0];
    const float* conv1_w  = (const float*)d_in[1];
    const float* bn1_s    = (const float*)d_in[2];
    const float* bn1_b    = (const float*)d_in[3];
    const float* conv2_w  = (const float*)d_in[4];
    const float* bn2_s    = (const float*)d_in[5];
    const float* bn2_b    = (const float*)d_in[6];
    const float* conv3_w  = (const float*)d_in[7];
    const float* bn3_s    = (const float*)d_in[8];
    const float* bn3_b    = (const float*)d_in[9];
    const float* lin1_w   = (const float*)d_in[10];
    const float* lin1_b   = (const float*)d_in[11];
    const float* lin2_w   = (const float*)d_in[12];
    const float* lin2_b   = (const float*)d_in[13];
    const float* aggw     = (const float*)d_in[14];
    const float* model_w  = (const float*)d_in[15];
    const float* model_b  = (const float*)d_in[16];
    const float* means    = (const float*)d_in[17];
    const float* stds     = (const float*)d_in[18];

    char* ws = (char*)d_ws;
    _Float16* h1   = (_Float16*)(ws + O_H1);
    _Float16* h2   = (_Float16*)(ws + O_H2);
    _Float16* wp2  = (_Float16*)(ws + O_WP2);
    _Float16* wp3  = (_Float16*)(ws + O_WP3);
    _Float16* wl1  = (_Float16*)(ws + O_WL1);
    _Float16* wl2  = (_Float16*)(ws + O_WL2);
    float*    wpm  = (float*)(ws + O_WPM);
    float*    S    = (float*)(ws + O_S);
    float*    feats= (float*)(ws + O_FEATS);
    float*    hid  = (float*)(ws + O_HID);
    float*    mem  = (float*)(ws + O_MEM);
    float*    outr = (float*)(ws + O_OUTR);
    float*    out  = (float*)d_out;

    hipMemsetAsync(feats, 0, 131072, stream);
    hipMemsetAsync(outr, 0, 8192, stream);

    // weight pre-packs (BN scale folded into conv B fragments)
    k_pack_conv  <<<72, 256, 0, stream>>>(conv2_w, bn2_s, wp2, 4, 1);
    k_pack_conv  <<<288, 256, 0, stream>>>(conv3_w, bn3_s, wp3, 8, 2);
    k_pack_linear<<<64, 256, 0, stream>>>(lin1_w, wl1, 128, 128);
    k_pack_linear<<<16, 256, 0, stream>>>(lin2_w, wl2, 24, 128);
    k_wp         <<<4704, 256, 0, stream>>>(model_w, stds, wpm);
    k_wsum       <<<24, 256, 0, stream>>>(model_w, S);

    // backbone
    k_conv1<<<12544, 256, 0, stream>>>(x, conv1_w, bn1_s, bn1_b, h1);
    k_conv2<<<dim3(49, 256), 128, 0, stream>>>(h1, wp2, bn2_b, h2);
    k_conv3<<<dim3(13, 256), 128, 0, stream>>>(h2, wp3, bn3_b, feats);
    k_lin1 <<<16, 256, 0, stream>>>(feats, wl1, lin1_b, hid);
    k_lin2 <<<16, 64, 0, stream>>>(hid, wl2, lin2_b, mem);

    // frozen sub-models + gating combine
    k_outputs<<<dim3(32, 12), 256, 0, stream>>>(x, wpm, outr);
    k_combine<<<1, 256, 0, stream>>>(mem, aggw, outr, model_b, means, stds, S, out);
}
